// Encoder_WordLstm_49555332661782
// MI455X (gfx1250) — matrix operand order repaired and checked by reading, not compiled, not run on hardware
//
#include <hip/hip_runtime.h>
#include <cmath>

// ---------------------------------------------------------------------------
// Types for CDNA5 WMMA (wave32): v_wmma_f32_16x16x32_bf16
// ---------------------------------------------------------------------------
typedef __bf16 bf16;
typedef __attribute__((ext_vector_type(16))) __bf16 bf16x16;
typedef __attribute__((ext_vector_type(8)))  float  f32x8;

static constexpr int BATCH  = 32;
static constexpr int SEQ    = 256;
static constexpr int HID    = 512;    // projection dim (LSTM input)
static constexpr int RDIM   = 512;    // LSTM hidden
static constexpr int INDIM  = 800;    // 2*(200+200)
static constexpr int NGATE  = 4 * RDIM; // 2048
static constexpr int ROWS   = SEQ * BATCH; // 8192
static constexpr int HLD    = RDIM + 16;   // padded LDS row of h (bf16)

__device__ __forceinline__ bf16 f2bf(float f) {
  union { float f; unsigned u; } v; v.f = f;
  unsigned r = v.u + 0x7FFFu + ((v.u >> 16) & 1u); // round-to-nearest-even
  union { unsigned short s; bf16 b; } o; o.s = (unsigned short)(r >> 16);
  return o.b;
}

__device__ __forceinline__ int flip_t(int t) {
  // full reversal except the middle pair (T even): 127->127, 128->128
  const int m = SEQ / 2;
  return (t == m - 1 || t == m) ? t : (SEQ - 1 - t);
}

__device__ __forceinline__ f32x8 vzero() {
  f32x8 z = {0.f, 0.f, 0.f, 0.f, 0.f, 0.f, 0.f, 0.f};
  return z;
}

// Branchless fast activations: v_exp_f32 + v_rcp_f32, no EXEC divergence.
__device__ __forceinline__ float fast_sigmoid(float x) {
  return __builtin_amdgcn_rcpf(1.f + __expf(-x));
}
__device__ __forceinline__ float fast_tanh(float x) {
  return 2.f * __builtin_amdgcn_rcpf(1.f + __expf(-2.f * x)) - 1.f;
}

// ---------------------------------------------------------------------------
// WMMA tile helpers — A/B use contiguous K (not the ISA table order); C/D per CDNA5 ISA 7.12.2 (wave32).
// All offset math in 32-bit int so global accesses use the SADDR+voffset
// form (uniform base in SGPRs, shared 32-bit offset VGPR).
// A 16x32 bf16 (MxK): lane m=lane&15, half=lane>>4; 16 contiguous K = 16*half.. (same as B)
// B 32x16 bf16 (KxN): lane n=lane&15, half=lane>>4; 16 contiguous K = 16*half..
// C/D 16x16 f32: lane n=lane&15, half=lane>>4; vgpr v holds row M = v + 8*half
// ---------------------------------------------------------------------------
__device__ __forceinline__ bf16x16 load_a_tile(const bf16* __restrict__ src,
                                               int ld, int row0, int k0, int lane) {
  const int m = lane & 15, half = lane >> 4;
  const int base = (row0 + m) * ld + k0 + half * 16;
  bf16x16 a;
#pragma unroll
  for (int v = 0; v < 8; ++v) {
    const int kb = 2 * v; // 0,2,4,...,14
    a[2 * v]     = src[base + kb];
    a[2 * v + 1] = src[base + kb + 1];
  }
  return a;
}

// B tile given a precomputed flat element index (16 contiguous bf16 = 32B).
__device__ __forceinline__ bf16x16 load_b_flat(const bf16* __restrict__ w,
                                               int idx) {
  bf16x16 b;
#pragma unroll
  for (int e = 0; e < 16; ++e) b[e] = w[idx + e]; // -> 2x global_load_b128
  return b;
}

// W row-major [N,K]; B tile (KxN) at (n0,k0): B[k][n] = W[n0+n][k0+k]
__device__ __forceinline__ bf16x16 load_b_tile(const bf16* __restrict__ w,
                                               int ld, int n0, int k0, int lane) {
  const int n = lane & 15, half = lane >> 4;
  return load_b_flat(w, (n0 + n) * ld + k0 + half * 16);
}

// Nontemporal f32 accumulator load (gate pre-activations are read once).
__device__ __forceinline__ f32x8 load_acc_f32_nt(const float* __restrict__ src,
                                                 int ld, int row0, int n0, int lane) {
  const int n = lane & 15, half = lane >> 4;
  f32x8 d;
#pragma unroll
  for (int v = 0; v < 8; ++v)
    d[v] = __builtin_nontemporal_load(&src[(row0 + v + 8 * half) * ld + n0 + n]);
  return d;
}

__device__ __forceinline__ f32x8 wmma_bf16(bf16x16 a, bf16x16 b, f32x8 c) {
  return __builtin_amdgcn_wmma_f32_16x16x32_bf16(false, a, false, b,
                                                 (short)0, c, false, false);
}

__device__ __forceinline__ f32x8 vsigmoid(f32x8 x) {
  f32x8 r;
#pragma unroll
  for (int e = 0; e < 8; ++e) r[e] = fast_sigmoid(x[e]);
  return r;
}
__device__ __forceinline__ f32x8 vtanh(f32x8 x) {
  f32x8 r;
#pragma unroll
  for (int e = 0; e < 8; ++e) r[e] = fast_tanh(x[e]);
  return r;
}

// ---------------------------------------------------------------------------
// Small prep kernels
// ---------------------------------------------------------------------------
__global__ void f32_to_bf16_kernel(const float* __restrict__ src,
                                   bf16* __restrict__ dst, int n) {
  int i = blockIdx.x * blockDim.x + threadIdx.x;
  if (i < n) dst[i] = f2bf(src[i]);
}

__global__ void bias_sum_kernel(const float* __restrict__ a,
                                const float* __restrict__ b,
                                float* __restrict__ out, int n) {
  int i = blockIdx.x * blockDim.x + threadIdx.x;
  if (i < n) out[i] = a[i] + b[i];
}

// Gather embeddings -> bf16 feature rows in (t,b) order.
// Right direction is gathered at flip(t) (flip is an involution; un-flipped at
// the final output store), so both LSTMs run in natural step order.
__global__ void gather_kernel(const int* __restrict__ ci,  const int* __restrict__ bli,
                              const int* __restrict__ bri, const int* __restrict__ sci,
                              const int* __restrict__ sbli,const int* __restrict__ sbri,
                              const float* __restrict__ cw, const float* __restrict__ scw,
                              const float* __restrict__ bw, const float* __restrict__ sbw,
                              bf16* __restrict__ xfl, bf16* __restrict__ xfr) {
  const int row = blockIdx.x;           // t*BATCH + b
  const int t = row / BATCH, b = row % BATCH;
  const int tl = t, tr = flip_t(t);
  for (int c = threadIdx.x; c < INDIM; c += blockDim.x) {
    const int seg = c / 200, off = c % 200;
    float vl, vr;
    if (seg == 0)      { vl = cw [ci  [b * SEQ + tl] * 200 + off];
                         vr = cw [ci  [b * SEQ + tr] * 200 + off]; }
    else if (seg == 1) { vl = scw[sci [b * SEQ + tl] * 200 + off];
                         vr = scw[sci [b * SEQ + tr] * 200 + off]; }
    else if (seg == 2) { vl = bw [bli [b * SEQ + tl] * 200 + off];
                         vr = bw [bri [b * SEQ + tr] * 200 + off]; }
    else               { vl = sbw[sbli[b * SEQ + tl] * 200 + off];
                         vr = sbw[sbri[b * SEQ + tr] * 200 + off]; }
    xfl[row * INDIM + c] = f2bf(vl);
    xfr[row * INDIM + c] = f2bf(vr);
  }
}

// ---------------------------------------------------------------------------
// Generic WMMA GEMM: C = op(A[MxK]bf16 @ W[NxK]^T bf16 + bias[N])
// Block = 8 waves (256 thr) -> 64x128 tile; each wave a 32x32 tile, K-major.
// ---------------------------------------------------------------------------
template <bool DO_TANH, bool OUT_BF16>
__global__ __launch_bounds__(256)
void gemm_wmma_kernel(const bf16* __restrict__ A, const bf16* __restrict__ W,
                      const float* __restrict__ bias, void* __restrict__ Cout,
                      int M, int N, int K) {
  const int wave = threadIdx.x >> 5, lane = threadIdx.x & 31;
  const int wm = wave >> 2, wn = wave & 3;
  const int row0 = blockIdx.x * 64 + wm * 32;
  const int n0   = blockIdx.y * 128 + wn * 32;

  f32x8 acc[2][2];
#pragma unroll
  for (int i = 0; i < 2; ++i)
#pragma unroll
    for (int j = 0; j < 2; ++j) acc[i][j] = vzero();

  for (int k0 = 0; k0 < K; k0 += 32) {
    if (k0 + 32 < K) { // speculative L2 prefetch of next weight panel
      __builtin_prefetch(&W[n0 * K + k0 + 32], 0, 1);
      __builtin_prefetch(&W[(n0 + 16) * K + k0 + 32], 0, 1);
    }
    bf16x16 a0 = load_a_tile(A, K, row0,      k0, lane);
    bf16x16 a1 = load_a_tile(A, K, row0 + 16, k0, lane);
    bf16x16 b0 = load_b_tile(W, K, n0,        k0, lane);
    bf16x16 b1 = load_b_tile(W, K, n0 + 16,   k0, lane);
    acc[0][0] = wmma_bf16(a0, b0, acc[0][0]);
    acc[0][1] = wmma_bf16(a0, b1, acc[0][1]);
    acc[1][0] = wmma_bf16(a1, b0, acc[1][0]);
    acc[1][1] = wmma_bf16(a1, b1, acc[1][1]);
  }

  const int n = lane & 15, half = lane >> 4;
#pragma unroll
  for (int mt = 0; mt < 2; ++mt)
#pragma unroll
    for (int nt = 0; nt < 2; ++nt)
#pragma unroll
      for (int v = 0; v < 8; ++v) {
        const int row = row0 + mt * 16 + v + 8 * half;
        const int col = n0 + nt * 16 + n;
        float val = acc[mt][nt][v] + bias[col];
        if (DO_TANH) val = fast_tanh(val);
        if (OUT_BF16) ((bf16*)Cout)[row * N + col] = f2bf(val);
        else __builtin_nontemporal_store(val, &((float*)Cout)[row * N + col]);
      }
}

// ---------------------------------------------------------------------------
// Recurrent LSTM: one block (32 waves, 1024 thr) per direction.
// gates_t[32,2048] = G_t (precomputed x@Wih^T + biases) + h_{t-1} @ Whh^T.
// Wave w owns hidden slice r0 = 16w and computes all 4 gate columns for it,
// so i,f,g,o and the cell state c stay in that wave's registers.
// h (32x512 bf16) lives in LDS; two wg barriers per step.
// Each gate's Whh block gets its own uniform (SGPR) base pointer so all 8
// B-loads per k-step share ONE 32-bit voffset VGPR (saddr addressing).
// ---------------------------------------------------------------------------
__global__ __launch_bounds__(1024)
void lstm_rec_kernel(const float* __restrict__ Gl, const float* __restrict__ Gr,
                     const bf16* __restrict__ whh_l, const bf16* __restrict__ whh_r,
                     float* __restrict__ out) {
  const int dir  = blockIdx.x;
  const float* G   = dir ? Gr : Gl;
  const bf16*  whh = dir ? whh_r : whh_l;
  // uniform per-gate weight bases (scalar adds, stay in SGPRs)
  const bf16* wg[4] = { whh, whh + RDIM * RDIM, whh + 2 * RDIM * RDIM,
                        whh + 3 * RDIM * RDIM };

  const int wave = threadIdx.x >> 5, lane = threadIdx.x & 31;
  const int r0 = wave * 16;
  const int n = lane & 15, half = lane >> 4;
  // shared per-lane B voffset (identical for all four gates)
  const int bidx0 = (r0 + n) * RDIM + half * 16;

  __shared__ bf16 hsh[BATCH * HLD];
  for (int i = threadIdx.x; i < BATCH * HLD; i += blockDim.x) hsh[i] = f2bf(0.f);
  __syncthreads();

  f32x8 c0 = vzero(), c1 = vzero(); // cell state, batch rows 0-15 / 16-31

  for (int t = 0; t < SEQ; ++t) {
    const float* Grow = G + t * (BATCH * NGATE); // < 2^29 elements: int math
    if (t + 1 < SEQ)
      __builtin_prefetch(&G[(t + 1) * (BATCH * NGATE) + r0], 0, 0);

    // init accumulators with precomputed input gates (+ biases), nontemporal
    f32x8 acc[2][4];
#pragma unroll
    for (int mt = 0; mt < 2; ++mt)
#pragma unroll
      for (int g = 0; g < 4; ++g)
        acc[mt][g] = load_acc_f32_nt(Grow, NGATE, mt * 16, g * RDIM + r0, lane);

    // recurrent GEMM: h_{t-1}[32x512] @ Whh^T slice (each Whh element read
    // exactly once per step across the 32 waves; stays resident in L2)
    for (int ks = 0; ks < RDIM / 32; ++ks) {
      const int k0 = ks * 32;
      bf16x16 a0 = load_a_tile(hsh, HLD, 0,  k0, lane);
      bf16x16 a1 = load_a_tile(hsh, HLD, 16, k0, lane);
      const int bidx = bidx0 + k0;
      bf16x16 bt[4];
#pragma unroll
      for (int g = 0; g < 4; ++g) bt[g] = load_b_flat(wg[g], bidx);
#pragma unroll
      for (int g = 0; g < 4; ++g) {
        acc[0][g] = wmma_bf16(a0, bt[g], acc[0][g]);
        acc[1][g] = wmma_bf16(a1, bt[g], acc[1][g]);
      }
    }

    __syncthreads(); // everyone done reading h_{t-1} from LDS

    const int tt_out = dir ? flip_t(t) : t;
#pragma unroll
    for (int mt = 0; mt < 2; ++mt) {
      f32x8 ig = vsigmoid(acc[mt][0]);
      f32x8 fg = vsigmoid(acc[mt][1]);
      f32x8 gg = vtanh   (acc[mt][2]);
      f32x8 og = vsigmoid(acc[mt][3]);
      f32x8& c = mt ? c1 : c0;
#pragma unroll
      for (int v = 0; v < 8; ++v) c[v] = fg[v] * c[v] + ig[v] * gg[v];
      f32x8 ct = vtanh(c);
#pragma unroll
      for (int v = 0; v < 8; ++v) {
        const float hv = og[v] * ct[v];
        const int bb = mt * 16 + v + 8 * half; // batch row
        hsh[bb * HLD + r0 + n] = f2bf(hv);
        __builtin_nontemporal_store(
            hv, &out[(bb * SEQ + tt_out) * (2 * RDIM) + dir * RDIM + r0 + n]);
      }
    }
    __syncthreads(); // h_t fully written before next step reads it
  }
}

// ---------------------------------------------------------------------------
// Host launcher
// ---------------------------------------------------------------------------
extern "C" void kernel_launch(void* const* d_in, const int* in_sizes, int n_in,
                              void* d_out, int out_size, void* d_ws, size_t ws_size,
                              hipStream_t stream) {
  (void)in_sizes; (void)n_in; (void)out_size; (void)ws_size;

  const int*   ci    = (const int*)  d_in[0];
  const int*   bli   = (const int*)  d_in[1];
  const int*   bri   = (const int*)  d_in[2];
  const int*   sci   = (const int*)  d_in[3];
  const int*   sbli  = (const int*)  d_in[4];
  const int*   sbri  = (const int*)  d_in[5];
  const float* cw    = (const float*)d_in[6];
  const float* scw   = (const float*)d_in[7];
  const float* bw    = (const float*)d_in[8];
  const float* sbw   = (const float*)d_in[9];
  const float* wlin  = (const float*)d_in[10];
  const float* blin  = (const float*)d_in[11];
  const float* wih_l = (const float*)d_in[12];
  const float* whh_l = (const float*)d_in[13];
  const float* bih_l = (const float*)d_in[14];
  const float* bhh_l = (const float*)d_in[15];
  const float* wih_r = (const float*)d_in[16];
  const float* whh_r = (const float*)d_in[17];
  const float* bih_r = (const float*)d_in[18];
  const float* bhh_r = (const float*)d_in[19];
  float* out = (float*)d_out;

  // ---- carve workspace (256B aligned) ----
  char* ws = (char*)d_ws;
  auto carve = [&](size_t bytes) -> void* {
    void* p = (void*)ws;
    ws += (bytes + 255) & ~(size_t)255;
    return p;
  };
  bf16* wlin_bf  = (bf16*)carve((size_t)HID * INDIM * 2);
  bf16* wih_l_bf = (bf16*)carve((size_t)NGATE * HID  * 2);
  bf16* whh_l_bf = (bf16*)carve((size_t)NGATE * RDIM * 2);
  bf16* wih_r_bf = (bf16*)carve((size_t)NGATE * HID  * 2);
  bf16* whh_r_bf = (bf16*)carve((size_t)NGATE * RDIM * 2);
  float* bias_l  = (float*)carve((size_t)NGATE * 4);
  float* bias_r  = (float*)carve((size_t)NGATE * 4);
  bf16* xfeat_l  = (bf16*)carve((size_t)ROWS * INDIM * 2);
  bf16* xfeat_r  = (bf16*)carve((size_t)ROWS * INDIM * 2);
  bf16* xl       = (bf16*)carve((size_t)ROWS * HID * 2);
  bf16* xr       = (bf16*)carve((size_t)ROWS * HID * 2);
  float* Gl      = (float*)carve((size_t)ROWS * NGATE * 4);
  float* Gr      = (float*)carve((size_t)ROWS * NGATE * 4);

  // ---- weight conversion to bf16 ----
  auto cvt = [&](const float* s, bf16* d, int nvals) {
    f32_to_bf16_kernel<<<(nvals + 255) / 256, 256, 0, stream>>>(s, d, nvals);
  };
  cvt(wlin,  wlin_bf,  HID * INDIM);
  cvt(wih_l, wih_l_bf, NGATE * HID);
  cvt(whh_l, whh_l_bf, NGATE * RDIM);
  cvt(wih_r, wih_r_bf, NGATE * HID);
  cvt(whh_r, whh_r_bf, NGATE * RDIM);
  bias_sum_kernel<<<(NGATE + 255) / 256, 256, 0, stream>>>(bih_l, bhh_l, bias_l, NGATE);
  bias_sum_kernel<<<(NGATE + 255) / 256, 256, 0, stream>>>(bih_r, bhh_r, bias_r, NGATE);

  // ---- gather embeddings (right pre-flipped) ----
  gather_kernel<<<ROWS, 256, 0, stream>>>(ci, bli, bri, sci, sbli, sbri,
                                          cw, scw, bw, sbw, xfeat_l, xfeat_r);

  // ---- projection: tanh(X @ Wlin^T + b) -> bf16 [8192,512] ----
  dim3 gproj(ROWS / 64, HID / 128);
  gemm_wmma_kernel<true, true><<<gproj, 256, 0, stream>>>(
      xfeat_l, wlin_bf, blin, (void*)xl, ROWS, HID, INDIM);
  gemm_wmma_kernel<true, true><<<gproj, 256, 0, stream>>>(
      xfeat_r, wlin_bf, blin, (void*)xr, ROWS, HID, INDIM);

  // ---- input gate pre-activations: G = X @ Wih^T + (bih+bhh) -> f32 ----
  dim3 ggate(ROWS / 64, NGATE / 128);
  gemm_wmma_kernel<false, false><<<ggate, 256, 0, stream>>>(
      xl, wih_l_bf, bias_l, (void*)Gl, ROWS, NGATE, HID);
  gemm_wmma_kernel<false, false><<<ggate, 256, 0, stream>>>(
      xr, wih_r_bf, bias_r, (void*)Gr, ROWS, NGATE, HID);

  // ---- sequential recurrence, one workgroup per direction ----
  lstm_rec_kernel<<<2, 1024, 0, stream>>>(Gl, Gr, whh_l_bf, whh_r_bf, out);
}
